// AttentionMechanism_29987461660983
// MI455X (gfx1250) — matrix operand order repaired and checked by reading, not compiled, not run on hardware
//
#include <hip/hip_runtime.h>

typedef __attribute__((ext_vector_type(16))) _Float16 v16h;
typedef __attribute__((ext_vector_type(8)))  _Float16 v8h;
typedef __attribute__((ext_vector_type(4)))  _Float16 v4h;
typedef __attribute__((ext_vector_type(8)))  float    v8f;

#define BM 128
#define BN 128
#define BK 64
#define LPAD 72   // padded LDS row (144 B): 16B aligned, bank-conflict free frag reads

#ifndef __has_builtin
#define __has_builtin(x) 0
#endif

#if __has_builtin(__builtin_amdgcn_global_load_async_to_lds_b128)
#define HAVE_ASYNC 1
#else
#define HAVE_ASYNC 0
#endif

#if HAVE_ASYNC
// Param 0 is 'int __vector(4) __device__ *' (AS1); param 1 is the AS3 pointer.
typedef int v4i_vs __attribute__((vector_size(16)));
typedef __attribute__((address_space(1))) v4i_vs* async_gp_t;
typedef __attribute__((address_space(3))) v4i_vs* async_lp_t;

__device__ __forceinline__ void async_cp16(const _Float16* g, _Float16* l) {
  __builtin_amdgcn_global_load_async_to_lds_b128(
      (async_gp_t)(void*)const_cast<_Float16*>(g),
      (async_lp_t)(void*)l, 0, 0);
}
__device__ __forceinline__ void wait_async() {
#if __has_builtin(__builtin_amdgcn_s_wait_asynccnt)
  __builtin_amdgcn_s_wait_asynccnt(0);
#else
  asm volatile("s_wait_asynccnt 0" ::: "memory");
#endif
}
#endif

// branchless tanh: tanh(x) = (e^{2x}-1)/(e^{2x}+1), clamped so e stays finite
__device__ __forceinline__ float fast_tanh(float x) {
  float xc = fminf(fmaxf(x, -10.0f), 10.0f);
  float e  = __expf(2.0f * xc);
  return (e - 1.0f) * __frcp_rn(e + 1.0f);
}

// ---------------------------------------------------------------------------
// f32 -> f16 conversion, 4 elements per thread (n must be a multiple of 4)
// ---------------------------------------------------------------------------
__global__ __launch_bounds__(256) void cvt_f32_f16(const float* __restrict__ in,
                                                   _Float16* __restrict__ out,
                                                   int n4) {
  int i = blockIdx.x * 256 + threadIdx.x;
  if (i < n4) {
    float4 v = ((const float4*)in)[i];
    v4h o = { (_Float16)v.x, (_Float16)v.y, (_Float16)v.z, (_Float16)v.w };
    ((v4h*)out)[i] = o;
  }
}

// ---------------------------------------------------------------------------
// WMMA GEMM: C[M,N] = op(A[M,K] * B[K,N])   (row-major, A/B f16)
// EPI == 0: C f32, plain store     EPI == 1: C f16, tanh(x + bias)
// Batched via blockIdx.z (element strides sA/sB/sC).
// Double-buffered LDS: next tile's async A-DMA + B global loads are issued
// before computing the current tile, so the DMA engine overlaps the 16 WMMAs.
// Block = 256 threads = 8 waves (4 in M x 2 in N); wave tile 32x64 = 2x4 WMMA.
// ---------------------------------------------------------------------------
template <int EPI>
__global__ __launch_bounds__(256) void wmma_gemm(
    const _Float16* __restrict__ A, const _Float16* __restrict__ Bm,
    void* __restrict__ C, int M, int N, int K,
    unsigned long long sA, unsigned long long sB, unsigned long long sC,
    const float* __restrict__ bias) {
  __shared__ alignas(16) _Float16 Alds[2][BM][LPAD];   // [m][k], rows padded
  __shared__ alignas(16) _Float16 Blds[2][BN][LPAD];   // [n][k] (transposed at fill)

  A  += (unsigned long long)blockIdx.z * sA;
  Bm += (unsigned long long)blockIdx.z * sB;

  const int tid  = threadIdx.x;
  const int lane = tid & 31;
  const int wave = tid >> 5;
  const int wm   = wave & 3;        // wave row (M): 0..3 -> 32 rows each
  const int wn   = wave >> 2;       // wave col (N): 0..1 -> 64 cols each
  const int l16  = lane & 15;
  const int hi   = lane >> 4;
  const int block_m = blockIdx.x * BM;
  const int block_n = blockIdx.y * BN;

  v8f acc[2][4] = {};
  v8h breg[4];

  // --- staging helpers -----------------------------------------------------
  auto stage_A = [&](int buf, int k0) {
#pragma unroll
    for (int i = 0; i < 4; ++i) {
      int chunk = tid + 256 * i;
      int r = chunk >> 3;
      int c = (chunk & 7) << 3;
      const _Float16* src = A + (unsigned long long)(block_m + r) * K + k0 + c;
#if HAVE_ASYNC
      async_cp16(src, &Alds[buf][r][c]);
#else
      *(v8h*)&Alds[buf][r][c] = *(const v8h*)src;
#endif
      if (i == 0 && k0 + BK < K) __builtin_prefetch(src + BK, 0, 3);
    }
  };
  auto load_B = [&](int k0) {
#pragma unroll
    for (int i = 0; i < 2; ++i) {
      int wrk = tid + 256 * i;            // 0..511
      int kp  = wrk >> 4;                 // K pair index 0..31
      int n0  = (wrk & 15) << 3;          // 0..120
      const _Float16* src = Bm + (unsigned long long)(k0 + 2 * kp) * N + block_n + n0;
      breg[2 * i]     = *(const v8h*)src;
      breg[2 * i + 1] = *(const v8h*)(src + N);
      if (i == 0 && k0 + BK < K) __builtin_prefetch(src + (unsigned long long)BK * N, 0, 3);
    }
  };
  auto store_B = [&](int buf) {
#pragma unroll
    for (int i = 0; i < 2; ++i) {
      int wrk = tid + 256 * i;
      int kp  = wrk >> 4;
      int n0  = (wrk & 15) << 3;
#pragma unroll
      for (int e = 0; e < 8; ++e) {
        union { _Float16 h[2]; unsigned int u; } p;
        p.h[0] = breg[2 * i][e];
        p.h[1] = breg[2 * i + 1][e];
        *(unsigned int*)&Blds[buf][n0 + e][2 * kp] = p.u;
      }
    }
  };
  auto compute = [&](int buf) {
#pragma unroll
    for (int ks = 0; ks < BK; ks += 32) {
      // B 32x16: lane = N (mod 16), K = slot + 16*(lane>=16), K contiguous in LDS
      v16h bfrag[4];
#pragma unroll
      for (int j = 0; j < 4; ++j) {
        int n = wn * 64 + j * 16 + l16;
        bfrag[j] = *(const v16h*)&Blds[buf][n][ks + 16 * hi];
      }
      // A 16x32: lane = M (mod 16); K = slot + 16*(lane>=16), K contiguous in LDS
      v16h afrag[2];
#pragma unroll
      for (int i = 0; i < 2; ++i) {
        int m = wm * 32 + i * 16 + l16;
        union { v16h v; v8h h[2]; } uf;
        uf.h[0] = *(const v8h*)&Alds[buf][m][ks + 16 * hi];
        uf.h[1] = *(const v8h*)&Alds[buf][m][ks + 16 * hi + 8];
        afrag[i] = uf.v;
      }
#pragma unroll
      for (int i = 0; i < 2; ++i)
#pragma unroll
        for (int j = 0; j < 4; ++j)
          acc[i][j] = __builtin_amdgcn_wmma_f32_16x16x32_f16(
              false, afrag[i], false, bfrag[j], (short)0, acc[i][j], false, false);
    }
  };

  // --- prologue: stage tile 0 into buffer 0 --------------------------------
  stage_A(0, 0);
  load_B(0);
  store_B(0);
#if HAVE_ASYNC
  wait_async();
#endif
  __syncthreads();

  // --- main loop: overlap next-tile staging with current-tile WMMAs --------
  const int nk = K / BK;
  for (int it = 0; it < nk; ++it) {
    int buf = it & 1;
    bool has_next = (it + 1 < nk);
    if (has_next) {
      stage_A(buf ^ 1, (it + 1) * BK);   // async DMA overlaps compute below
      load_B((it + 1) * BK);             // global loads in flight during compute
    }
    compute(buf);
    if (has_next) {
      store_B(buf ^ 1);
#if HAVE_ASYNC
      wait_async();
#endif
      __syncthreads();
    }
  }

  // --- epilogue: C/D layout lane = N (mod 16), VGPR r -> M = r + 8*(lane/16) ---
  const float bv = (EPI == 1) ? *bias : 0.0f;
#pragma unroll
  for (int i = 0; i < 2; ++i) {
#pragma unroll
    for (int j = 0; j < 4; ++j) {
      int n  = block_n + wn * 64 + j * 16 + l16;
      int mb = block_m + wm * 32 + i * 16 + hi * 8;
#pragma unroll
      for (int r = 0; r < 8; ++r) {
        unsigned long long idx =
            (unsigned long long)blockIdx.z * sC +
            (unsigned long long)(mb + r) * N + n;
        float v = acc[i][j][r];
        if (EPI == 1)
          ((_Float16*)C)[idx] = (_Float16)fast_tanh(v + bv);
        else
          ((float*)C)[idx] = v;
      }
    }
  }
}

// ---------------------------------------------------------------------------
// Softmax over L (dim 1) of scores[B, L, SEQ], written TRANSPOSED as
// at[B, SEQ, L] in f16. One block = one batch x 32 columns (s values).
// ---------------------------------------------------------------------------
__global__ __launch_bounds__(256) void softmax_cols(
    const float* __restrict__ sc, _Float16* __restrict__ at) {
  const int L = 512, SEQ = 512;
  const int b  = blockIdx.y;
  const int s0 = blockIdx.x * 32;
  const int t  = threadIdx.x;
  const int c  = t & 31;   // column within the 32-wide tile
  const int g  = t >> 5;   // 0..7 row group

  __shared__ float red[8][32];
  __shared__ float colmax[32];
  __shared__ float colsum[32];
  __shared__ alignas(16) _Float16 stage[32][72];

  const float* col = sc + (unsigned long long)b * L * SEQ + s0;

  // pass 1: column max (reads coalesced across c)
  float m = -3.4e38f;
  for (int l = g; l < L; l += 8) m = fmaxf(m, col[(unsigned long long)l * SEQ + c]);
  red[g][c] = m;
  __syncthreads();
  if (t < 32) {
    float mm = red[0][t];
#pragma unroll
    for (int i = 1; i < 8; ++i) mm = fmaxf(mm, red[i][t]);
    colmax[t] = mm;
  }
  __syncthreads();

  // pass 2: column sum of exp
  const float mv = colmax[c];
  float s = 0.f;
  for (int l = g; l < L; l += 8) s += __expf(col[(unsigned long long)l * SEQ + c] - mv);
  red[g][c] = s;
  __syncthreads();
  if (t < 32) {
    float ss = 0.f;
#pragma unroll
    for (int i = 0; i < 8; ++i) ss += red[i][t];
    colsum[t] = ss;
  }
  __syncthreads();
  const float inv = 1.0f / colsum[c];

  // pass 3: emit normalized values transposed, staged in LDS for coalesced writes
  _Float16* outb = at + ((unsigned long long)b * SEQ + s0) * L;
  for (int l0 = 0; l0 < L; l0 += 64) {
#pragma unroll
    for (int ii = 0; ii < 8; ++ii) {
      int lo = g + 8 * ii;
      float v = __expf(col[(unsigned long long)(l0 + lo) * SEQ + c] - mv) * inv;
      stage[c][lo] = (_Float16)v;
    }
    __syncthreads();
    {
      int c2 = t >> 3, j = t & 7;
      v8h vv = *(const v8h*)&stage[c2][j * 8];
      *(v8h*)&outb[(unsigned long long)c2 * L + l0 + j * 8] = vv;
    }
    __syncthreads();
  }
}

// ---------------------------------------------------------------------------
extern "C" void kernel_launch(void* const* d_in, const int* in_sizes, int n_in,
                              void* d_out, int out_size, void* d_ws, size_t ws_size,
                              hipStream_t stream) {
  constexpr int Bz = 64, L = 512, IN = 1024, HID = 1024, SEQ = 512;

  const float* h = (const float*)d_in[0];   // [B, L, IN]
  const float* w = (const float*)d_in[1];   // [IN, HID]
  const float* u = (const float*)d_in[2];   // [HID, SEQ]
  const float* b = (const float*)d_in[3];   // [1]
  float* out = (float*)d_out;               // [B, SEQ, IN]

  char* ws = (char*)d_ws;
  const size_t MB = (size_t)1 << 20;
  _Float16* h16 = (_Float16*)(ws + 0 * MB);     // 64 MB
  _Float16* w16 = (_Float16*)(ws + 64 * MB);    //  2 MB
  _Float16* u16 = (_Float16*)(ws + 66 * MB);    //  1 MB
  _Float16* a3  = (_Float16*)(ws + 67 * MB);    // 64 MB  tanh(h*w+b) f16
  float*    sc  = (float*)   (ws + 131 * MB);   // 64 MB  scores f32
  _Float16* at  = (_Float16*)(ws + 195 * MB);   // 32 MB  softmax, transposed [B,SEQ,L]

  // 1) precision demotion to f16 for the matrix units
  cvt_f32_f16<<<(Bz * L * IN) / 4 / 256, 256, 0, stream>>>(h, h16, (Bz * L * IN) / 4);
  cvt_f32_f16<<<(IN * HID) / 4 / 256, 256, 0, stream>>>(w, w16, (IN * HID) / 4);
  cvt_f32_f16<<<(HID * SEQ) / 4 / 256, 256, 0, stream>>>(u, u16, (HID * SEQ) / 4);

  // 2) a3 = tanh(h @ w + b) : [B*L, HID], f16 out
  {
    dim3 grid((Bz * L) / BM, HID / BN, 1);
    wmma_gemm<1><<<grid, 256, 0, stream>>>(h16, w16, (void*)a3,
                                           Bz * L, HID, IN, 0ull, 0ull, 0ull, b);
  }
  // 3) scores = a3 @ u : [B*L, SEQ], f32 out
  {
    dim3 grid((Bz * L) / BM, SEQ / BN, 1);
    wmma_gemm<0><<<grid, 256, 0, stream>>>(a3, u16, (void*)sc,
                                           Bz * L, SEQ, HID, 0ull, 0ull, 0ull, nullptr);
  }
  // 4) softmax over L, write transposed at[B, SEQ, L] f16
  {
    dim3 grid(SEQ / 32, Bz, 1);
    softmax_cols<<<grid, 256, 0, stream>>>(sc, at);
  }
  // 5) S[b] = at[b] @ h[b] : batched [SEQ, IN] = [SEQ, L] x [L, IN], f32 out
  {
    dim3 grid(SEQ / BM, IN / BN, Bz);
    wmma_gemm<0><<<grid, 256, 0, stream>>>(
        at, h16, (void*)out, SEQ, IN, L,
        (unsigned long long)SEQ * L, (unsigned long long)L * IN,
        (unsigned long long)SEQ * IN, nullptr);
  }
}
